// InfoNCE_47794396070560
// MI455X (gfx1250) — matrix operand order repaired and checked by reading, not compiled, not run on hardware
//
#include <hip/hip_runtime.h>
#include <math.h>

typedef __attribute__((ext_vector_type(16))) _Float16 v16h;
typedef __attribute__((ext_vector_type(8)))  float    v8f;

#define NTOT 1024
#define HDIM 64

// Workspace layout (bytes):
//   [0      .. 4095 ]  float lse[1024]   (per-row logsumexp)
//   [4096   .. 8191 ]  float t0[1024]    (diagonal T1[i,i] == T0[i])
//   [8192   .. +128K]  _Float16 Aswz[1024*64]  (zc@W1c, WMMA-A-fragment swizzled)
//   [139264 .. +256K]  float Bf[1024*64]       (zd@W1d, row-major f32)

// ---------------------------------------------------------------------------
// Kernel 1: per-sample first-layer partials.
//   A[j][h] = sum_k zc[j][k] * W1[k][h]        -> stored f16, fragment-swizzled
//   B[j][h] = sum_k zd[j][k] * W1[64+k][h]     -> stored f32, row-major
// A-fragment mapping (16-bit A, 16x32, ISA 7.12.2):
//   lane L, elem e, chunk c  <->  row = L&15, k = c*32 + ((e>>3)<<4) + ((L>>4)<<3) + (e&7)
// ---------------------------------------------------------------------------
__global__ void __launch_bounds__(64) precompute_kernel(
    const float* __restrict__ zc, const float* __restrict__ zd,
    const float* __restrict__ W1,
    _Float16* __restrict__ Aswz, float* __restrict__ Bf) {
  __shared__ float szc[64], szd[64];
  const int j = blockIdx.x;
  const int h = threadIdx.x;
  szc[h] = zc[j * 64 + h];
  szd[h] = zd[j * 64 + h];
  __syncthreads();
  float a = 0.f, b = 0.f;
#pragma unroll 8
  for (int k = 0; k < 64; ++k) {
    a = fmaf(szc[k], W1[k * 64 + h], a);
    b = fmaf(szd[k], W1[(64 + k) * 64 + h], b);
  }
  Bf[j * 64 + h] = b;
  // scatter a = A[j][h] into the WMMA A-fragment layout
  const int t = j >> 4, m = j & 15;          // j-tile, row-in-tile
  const int c = h >> 5, kk = h & 31;         // k-chunk, k-in-chunk
  const int e = ((kk >> 4) & 1) * 8 + (kk & 7);
  const int L = ((kk >> 3) & 1) * 16 + m;
  Aswz[(((t * 2 + c) * 32) + L) * 16 + e] = (_Float16)a;
}

// ---------------------------------------------------------------------------
// Kernel 2: pairwise critic + row-wise logsumexp. One block (4 waves) per i.
// Each wave handles 16 j-tiles of 16 pairs; per tile 8x v_wmma_f32_16x16x32_f16.
// ---------------------------------------------------------------------------
__global__ void __launch_bounds__(128, 1) pairwise_kernel(
    const _Float16* __restrict__ Aswz, const float* __restrict__ Bf,
    const float* __restrict__ b1, const float* __restrict__ W2,
    const float* __restrict__ b2, const float* __restrict__ Wo,
    const float* __restrict__ bo,
    float* __restrict__ lse_out, float* __restrict__ t0_out) {
  __shared__ _Float16 sW2[4096];   // 8 B-fragments of W2 (f16)
  __shared__ float    sBi[1024];   // (B[i] + b1) swizzled to A-fragment order
  __shared__ float    sT1[1024];   // T1[i, 0..1023]
  __shared__ float    red[128];

  const int tid  = threadIdx.x;
  const int i    = blockIdx.x;
  const int lane = tid & 31;
  const int wave = tid >> 5;

  // Stage W2 into LDS in B-fragment order (16-bit B, 32x16 per fragment):
  //   lane n, elem e, chunk c, n-tile nt  <->  k = c*32 + ((e>>3)<<4) + ((n>>4)<<3) + (e&7),
  //   col = nt*16 + (n&15).  Fragment index q = nt*2 + c.
  for (int idx = tid; idx < 4096; idx += 128) {
    const int q  = idx >> 9;
    const int ln = (idx >> 4) & 31;
    const int e  = idx & 15;
    const int nt = q >> 1, c = q & 1;
    const int k   = c * 32 + ((e >> 3) << 4) + ((ln >> 4) << 3) + (e & 7);
    const int col = nt * 16 + (ln & 15);
    sW2[idx] = (_Float16)W2[k * 64 + col];
  }
  // (B[i] + b1) mapped to the A-fragment element order so the h1 build is
  // a straight vector add against the preswizzled A tile.
  for (int idx = tid; idx < 1024; idx += 128) {
    const int c = idx >> 9;
    const int L = (idx >> 4) & 31;
    const int e = idx & 15;
    const int k = c * 32 + ((e >> 3) << 4) + ((L >> 4) << 3) + (e & 7);
    sBi[idx] = Bf[i * 64 + k] + b1[k];
  }
  __syncthreads();

  // Pull the 8 W2 B-fragments into registers (reused across all 16 tiles).
  v16h bfr[8];
#pragma unroll
  for (int q = 0; q < 8; ++q) bfr[q] = *(const v16h*)&sW2[q * 512 + lane * 16];

  // Per-lane epilogue constants: col = f*16 + (lane&15)  (C/D layout, ISA 7.12.2)
  float b2v[4], wov[4];
#pragma unroll
  for (int f = 0; f < 4; ++f) {
    const int col = f * 16 + (lane & 15);
    b2v[f] = b2[col];
    wov[f] = Wo[col];
  }
  const float bov = bo[0];

  for (int tt = 0; tt < 16; ++tt) {
    const int t = wave * 16 + tt;              // j-tile: j0 = t*16
    if (tt < 15)                               // prefetch next tile's A fragments
      __builtin_prefetch(&Aswz[(((t + 1) * 2) * 32 + lane) * 16], 0, 0);

    // Build h1 A-fragments: relu(A[j] + B[i] + b1), f32 add then cvt to f16.
    v16h afr[2];
#pragma unroll
    for (int c = 0; c < 2; ++c) {
      const v16h  araw = *(const v16h*)&Aswz[((t * 2 + c) * 32 + lane) * 16];
      const float* bi  = &sBi[c * 512 + lane * 16];
      v16h h;
#pragma unroll
      for (int e = 0; e < 16; ++e) {
        const float v = (float)araw[e] + bi[e];
        h[e] = (_Float16)fmaxf(v, 0.f);
      }
      afr[c] = h;
    }

    // h2_pre(16x64) = h1(16x64) @ W2(64x64): 4 N-tiles x 2 K-chunks = 8 WMMA.
    v8f acc[4];
#pragma unroll
    for (int f = 0; f < 4; ++f) {
      v8f cz = {};
      cz = __builtin_amdgcn_wmma_f32_16x16x32_f16(false, afr[0], false, bfr[f * 2 + 0],
                                                  (short)0, cz, false, false);
      cz = __builtin_amdgcn_wmma_f32_16x16x32_f16(false, afr[1], false, bfr[f * 2 + 1],
                                                  (short)0, cz, false, false);
      acc[f] = cz;
    }

    // Epilogue: T1 = relu(h2_pre + b2) . Wo + bo for the 16 pairs.
    // D layout: lane n, vgpr r -> row m = r + 8*(n>>4), col = 16*f + (n&15).
    float part[8];
#pragma unroll
    for (int r = 0; r < 8; ++r) {
      float s = 0.f;
#pragma unroll
      for (int f = 0; f < 4; ++f) s += fmaxf(acc[f][r] + b2v[f], 0.f) * wov[f];
#pragma unroll
      for (int mk = 1; mk < 16; mk <<= 1) s += __shfl_xor(s, mk, 32);
      part[r] = s;
    }
    if ((lane & 15) == 0) {
      const int mbase = (lane >> 4) * 8;
#pragma unroll
      for (int r = 0; r < 8; ++r) sT1[t * 16 + mbase + r] = part[r] + bov;
    }
  }
  __syncthreads();

  // Row-wise logsumexp over sT1[0..1023] (two-pass, LDS tree reduce).
  float m = -3.4e38f;
#pragma unroll
  for (int s = 0; s < 8; ++s) m = fmaxf(m, sT1[tid + 128 * s]);
  red[tid] = m;
  __syncthreads();
  for (int off = 64; off > 0; off >>= 1) {
    if (tid < off) red[tid] = fmaxf(red[tid], red[tid + off]);
    __syncthreads();
  }
  const float M = red[0];
  __syncthreads();
  float se = 0.f;
#pragma unroll
  for (int s = 0; s < 8; ++s) se += expf(sT1[tid + 128 * s] - M);
  red[tid] = se;
  __syncthreads();
  for (int off = 64; off > 0; off >>= 1) {
    if (tid < off) red[tid] += red[tid + off];
    __syncthreads();
  }
  if (tid == 0) {
    lse_out[i] = M + logf(red[0]);
    t0_out[i]  = sT1[i];          // diagonal: T1[i,i] == T0[i]
  }
}

// ---------------------------------------------------------------------------
// Kernel 3: final scalar.
//   out = mean(T0) - (mean(lse) - log(N))
// ---------------------------------------------------------------------------
__global__ void __launch_bounds__(256) finalize_kernel(
    const float* __restrict__ lse, const float* __restrict__ t0,
    float* __restrict__ out) {
  __shared__ float r1[256], r2[256];
  const int tid = threadIdx.x;
  float s1 = 0.f, s2 = 0.f;
  for (int k = tid; k < NTOT; k += 256) { s1 += lse[k]; s2 += t0[k]; }
  r1[tid] = s1; r2[tid] = s2;
  __syncthreads();
  for (int off = 128; off > 0; off >>= 1) {
    if (tid < off) { r1[tid] += r1[tid + off]; r2[tid] += r2[tid + off]; }
    __syncthreads();
  }
  if (tid == 0)
    out[0] = r2[0] / (float)NTOT - (r1[0] / (float)NTOT - logf((float)NTOT));
}

extern "C" void kernel_launch(void* const* d_in, const int* in_sizes, int n_in,
                              void* d_out, int out_size, void* d_ws, size_t ws_size,
                              hipStream_t stream) {
  (void)in_sizes; (void)n_in; (void)out_size; (void)ws_size;
  const float* zc = (const float*)d_in[0];
  const float* zd = (const float*)d_in[1];
  const float* W1 = (const float*)d_in[2];
  const float* b1 = (const float*)d_in[3];
  const float* W2 = (const float*)d_in[4];
  const float* b2 = (const float*)d_in[5];
  const float* Wo = (const float*)d_in[6];
  const float* bo = (const float*)d_in[7];
  float* out = (float*)d_out;

  char* ws = (char*)d_ws;
  float*    lse_arr = (float*)ws;                         // 1024 f32
  float*    t0_arr  = (float*)(ws + 4096);                // 1024 f32
  _Float16* Aswz    = (_Float16*)(ws + 8192);             // 1024*64 f16
  float*    Bf      = (float*)(ws + 8192 + NTOT * HDIM * 2);  // 1024*64 f32

  precompute_kernel<<<NTOT, 64, 0, stream>>>(zc, zd, W1, Aswz, Bf);
  pairwise_kernel<<<NTOT, 128, 0, stream>>>(Aswz, Bf, b1, W2, b2, Wo, bo,
                                            lse_arr, t0_arr);
  finalize_kernel<<<1, 256, 0, stream>>>(lse_arr, t0_arr, out);
}
